// MultiScaleRetention_25074019074211
// MI455X (gfx1250) — matrix operand order repaired and checked by reading, not compiled, not run on hardware
//
#include <hip/hip_runtime.h>
#include <math.h>

// ---------------- problem constants ----------------
constexpr int Bc = 2, Tc = 2048, Ec = 1024, Hc = 4;
constexpr int Dk = 256, Dv = 512;
constexpr int NT = Bc * Tc;          // 4096 tokens
constexpr float EPSc = 1e-5f;

// ---------------- WMMA types ----------------
typedef __attribute__((ext_vector_type(16))) __bf16 v16bf;
typedef __attribute__((ext_vector_type(8)))  __bf16 v8bf;
typedef __attribute__((ext_vector_type(8)))  float  v8f;

#define WMMA_BF16(a, b, c) \
  __builtin_amdgcn_wmma_f32_16x16x32_bf16(false, (a), false, (b), (short)0, (c), false, false)

// A-fragment (16x32 bf16, row-major source, row stride given by caller):
// lane-half lh: K = lh*8..lh*8+7 in elems [0..7], K = 16+lh*8.. in elems [8..15]
static __device__ __forceinline__ v16bf ldfragA(const __bf16* row, int lh) {
  v8bf lo = *(const v8bf*)(row + lh * 8);
  v8bf hi = *(const v8bf*)(row + lh * 8 + 16);
  return __builtin_shufflevector(lo, hi, 0,1,2,3,4,5,6,7,8,9,10,11,12,13,14,15);
}
// B-fragment (32x16 bf16 fed from row-major N x K storage):
// lane-half lh: K = lh*8..lh*8+7 in elems [0..7], K = 16+lh*8.. in elems [8..15]
static __device__ __forceinline__ v16bf ldfragB(const __bf16* row, int lh) {
  v8bf lo = *(const v8bf*)(row + lh * 8);
  v8bf hi = *(const v8bf*)(row + lh * 8 + 16);
  return __builtin_shufflevector(lo, hi, 0,1,2,3,4,5,6,7,8,9,10,11,12,13,14,15);
}

// ---------------- conversion kernels ----------------
__global__ __launch_bounds__(256) void cvt_kernel(const float* __restrict__ s,
                                                  __bf16* __restrict__ d, int n) {
  int i = blockIdx.x * 256 + threadIdx.x;
  if (i < n) d[i] = (__bf16)s[i];
}

// W (rows x cols, row-major f32)  ->  Wt (cols x rows, row-major bf16)
__global__ __launch_bounds__(256) void cvtT_kernel(const float* __restrict__ s,
                                                   __bf16* __restrict__ d,
                                                   int rows, int cols) {
  int i = blockIdx.x * 256 + threadIdx.x;
  if (i < rows * cols) {
    int r = i / cols, c = i - r * cols;
    d[(size_t)c * rows + r] = (__bf16)s[i];
  }
}

// ---------------- tiled WMMA GEMM:  C(MxN) = A(MxK) * Bt(NxK)^T ----------------
constexpr int BM = 128, BN = 128, BK = 32, LDT = 40;  // 8-elem pad per LDS row

template <typename OutT>
__global__ __launch_bounds__(256) void gemm_bf16_kernel(
    const __bf16* __restrict__ A, const __bf16* __restrict__ Bt,
    OutT* __restrict__ C, int M, int N, int K) {
  __shared__ __align__(16) __bf16 As[BM * LDT];
  __shared__ __align__(16) __bf16 Bs[BN * LDT];

  const int tid  = threadIdx.x;
  const int lane = tid & 31;
  const int wave = tid >> 5;
  const int wm = wave >> 2;       // 0..1  -> 64 rows of M
  const int wn = wave & 3;        // 0..3  -> 32 cols of N
  const int lm = lane & 15;
  const int lh = lane >> 4;

  const int m0 = blockIdx.y * BM;
  const int n0 = blockIdx.x * BN;

  v8f zero = {};
  v8f acc[4][2];
#pragma unroll
  for (int i = 0; i < 4; ++i)
#pragma unroll
    for (int j = 0; j < 2; ++j) acc[i][j] = zero;

  for (int kb = 0; kb < K; kb += BK) {
    __syncthreads();
#pragma unroll
    for (int j = 0; j < 2; ++j) {
      int c   = tid + j * 256;          // 512 16-byte chunks per tile
      int row = c >> 2;
      int col = (c & 3) * 8;
      const __bf16* ga = A  + (size_t)(m0 + row) * K + kb + col;
      const __bf16* gb = Bt + (size_t)(n0 + row) * K + kb + col;
      *(v8bf*)(&As[row * LDT + col]) = *(const v8bf*)ga;
      *(v8bf*)(&Bs[row * LDT + col]) = *(const v8bf*)gb;
      if (kb + BK < K) {                // CDNA5 global_prefetch of next K-slab
        __builtin_prefetch(ga + BK, 0, 3);
        __builtin_prefetch(gb + BK, 0, 3);
      }
    }
    __syncthreads();

    v16bf af[4];
#pragma unroll
    for (int mt = 0; mt < 4; ++mt)
      af[mt] = ldfragA(&As[(wm * 64 + mt * 16 + lm) * LDT], lh);
    v16bf bfr[2];
#pragma unroll
    for (int nt = 0; nt < 2; ++nt)
      bfr[nt] = ldfragB(&Bs[(wn * 32 + nt * 16 + lm) * LDT], lh);
#pragma unroll
    for (int mt = 0; mt < 4; ++mt)
#pragma unroll
      for (int nt = 0; nt < 2; ++nt)
        acc[mt][nt] = WMMA_BF16(af[mt], bfr[nt], acc[mt][nt]);
  }

  // D layout: element i of v8f -> M = i + 8*lh, N = lm
#pragma unroll
  for (int mt = 0; mt < 4; ++mt)
#pragma unroll
    for (int nt = 0; nt < 2; ++nt)
#pragma unroll
      for (int i = 0; i < 8; ++i) {
        int r  = m0 + wm * 64 + mt * 16 + i + lh * 8;
        int cc = n0 + wn * 32 + nt * 16 + lm;
        C[(size_t)r * N + cc] = (OutT)acc[mt][nt][i];
      }
}

// ---------------- rotary + q-scale; (B,T,H,Dk) -> (B,H,T,Dk) ----------------
__global__ __launch_bounds__(256) void rope_kernel(const __bf16* __restrict__ qp,
                                                   const __bf16* __restrict__ kp,
                                                   __bf16* __restrict__ qo,
                                                   __bf16* __restrict__ ko) {
  int idx = blockIdx.x * 256 + threadIdx.x;   // B*T*H*(Dk/2) = 2^21
  const int half = Dk / 2;                    // 128
  int f = idx & (half - 1);
  int h = (idx >> 7) & (Hc - 1);
  int t = (idx >> 9) & (Tc - 1);
  int b = idx >> 20;
  // inv_freq = 10000^(-2f/Dk) ; log2(10000) = 13.2877123795
  float inv = exp2f(-((float)(2 * f) / (float)Dk) * 13.287712379549449f);
  float ang = (float)t * inv;
  float cs = cosf(ang), sn = sinf(ang);

  size_t ib = ((size_t)(b * Tc + t)) * Ec + (size_t)h * Dk;
  float q1 = (float)qp[ib + f], q2 = (float)qp[ib + f + half];
  float k1 = (float)kp[ib + f], k2 = (float)kp[ib + f + half];
  size_t ob = (((size_t)b * Hc + h) * Tc + t) * Dk;
  const float sc = 0.0625f;                   // Dk^-0.5
  qo[ob + f]        = (__bf16)((q1 * cs - q2 * sn) * sc);
  qo[ob + f + half] = (__bf16)((q2 * cs + q1 * sn) * sc);
  ko[ob + f]        = (__bf16)(k1 * cs - k2 * sn);
  ko[ob + f + half] = (__bf16)(k2 * cs + k1 * sn);
}

// ---------------- V: (B,T,H,Dv) -> V^T (B,H,Dv,T) ----------------
__global__ __launch_bounds__(256) void vtrans_kernel(const __bf16* __restrict__ vp,
                                                     __bf16* __restrict__ vt) {
  int idx = blockIdx.x * 256 + threadIdx.x;   // 2^23
  int dv = idx & (Dv - 1);
  int h  = (idx >> 9) & (Hc - 1);
  int t  = (idx >> 11) & (Tc - 1);
  int b  = idx >> 22;
  vt[(((size_t)b * Hc + h) * Dv + dv) * Tc + t] = vp[idx];
}

// ---------------- retention: o = (QK^T ∘ decay) V ----------------
// one wave per (16 q-rows, 256 dv-cols, head)
__global__ __launch_bounds__(32) void retention_kernel(
    const __bf16* __restrict__ qr,   // (B,H,T,Dk)
    const __bf16* __restrict__ kr,   // (B,H,T,Dk)
    const __bf16* __restrict__ vt,   // (B,H,Dv,T)
    float* __restrict__ o) {         // (B,T,H,Dv)
  const int lane = threadIdx.x & 31;
  const int lm = lane & 15;
  const int lh = lane >> 4;
  const int t0  = blockIdx.x * 16;
  const int dv0 = blockIdx.y * 256;
  const int bh  = blockIdx.z;
  const int h = bh & (Hc - 1);
  const int b = bh >> 2;

  const float lg = log2f(1.0f - exp2f(-5.0f - (float)h));  // log2(gamma_h)

  const __bf16* Q  = qr + (size_t)bh * Tc * Dk;
  const __bf16* Kp = kr + (size_t)bh * Tc * Dk;
  const __bf16* V  = vt + (size_t)bh * Dv * Tc;

  __shared__ __align__(16) __bf16 Ps[16 * LDT];

  v8f zero = {};
  v8f acc[16];
#pragma unroll
  for (int i = 0; i < 16; ++i) acc[i] = zero;

  const __bf16* qrow = Q + (size_t)(t0 + lm) * Dk;

  for (int s0 = 0; s0 < t0 + 16; s0 += 32) {
    // ---- scores P(16 x 32) = Q(16xDk) @ K(s-block)^T ----
    v8f p0 = zero, p1 = zero;
#pragma unroll
    for (int dk = 0; dk < Dk; dk += 32) {
      v16bf af = ldfragA(qrow + dk, lh);
      v16bf b0 = ldfragB(Kp + (size_t)(s0 + lm) * Dk + dk, lh);
      v16bf b1 = ldfragB(Kp + (size_t)(s0 + 16 + lm) * Dk + dk, lh);
      p0 = WMMA_BF16(af, b0, p0);
      p1 = WMMA_BF16(af, b1, p1);
    }

    // ---- causal decay gamma^(t-s) on D-fragment, spill to LDS as A layout ----
#pragma unroll
    for (int i = 0; i < 8; ++i) {
      int t = t0 + i + lh * 8;
      int sA = s0 + lm;
      int sB = s0 + 16 + lm;
      int dA = t - sA, dB = t - sB;
      float wA = (dA >= 0) ? exp2f(lg * (float)dA) : 0.0f;
      float wB = (dB >= 0) ? exp2f(lg * (float)dB) : 0.0f;
      Ps[(i + lh * 8) * LDT + lm]      = (__bf16)(p0[i] * wA);
      Ps[(i + lh * 8) * LDT + 16 + lm] = (__bf16)(p1[i] * wB);
    }
    asm volatile("s_wait_dscnt 0x0" ::: "memory");
    v16bf pfrag = ldfragA(&Ps[lm * LDT], lh);
    asm volatile("s_wait_dscnt 0x0" ::: "memory");

    // ---- o += P @ V ;  V^T rows give contiguous-K B fragments ----
#pragma unroll
    for (int nt = 0; nt < 16; ++nt) {
      const __bf16* vrow = V + (size_t)(dv0 + nt * 16 + lm) * Tc + s0;
      v16bf bf = ldfragB(vrow, lh);
      acc[nt] = WMMA_BF16(pfrag, bf, acc[nt]);
    }
  }

#pragma unroll
  for (int nt = 0; nt < 16; ++nt)
#pragma unroll
    for (int i = 0; i < 8; ++i) {
      int t  = t0 + i + lh * 8;
      int dv = dv0 + nt * 16 + lm;
      o[(((size_t)b * Tc + t) * Hc + h) * Dv + dv] = acc[nt][i];
    }
}

// ---------------- RMS groupnorm over Dv, * gn_weight, * gate ----------------
__global__ __launch_bounds__(256) void gn_gate_kernel(const float* __restrict__ o,
                                                      const __bf16* __restrict__ g,
                                                      const float* __restrict__ gnw,
                                                      __bf16* __restrict__ og) {
  int bt  = blockIdx.x;                 // token
  int tid = threadIdx.x;
  const float*  orow = o  + (size_t)bt * (Hc * Dv);
  const __bf16* grow = g  + (size_t)bt * (Hc * Dv);
  __bf16*       wrow = og + (size_t)bt * (Hc * Dv);

  __shared__ float red[256];
  float vals[8];
  int base = tid * 8;                   // 64 threads per head group of 512
  float ss = 0.0f;
#pragma unroll
  for (int i = 0; i < 8; ++i) { vals[i] = orow[base + i]; ss += vals[i] * vals[i]; }
  red[tid] = ss;
  __syncthreads();
  for (int off = 32; off > 0; off >>= 1) {
    if ((tid & 63) < off) red[tid] += red[tid + off];
    __syncthreads();
  }
  float mean = red[(tid >> 6) << 6] * (1.0f / (float)Dv);
  float rinv = rsqrtf(mean + EPSc);
#pragma unroll
  for (int i = 0; i < 8; ++i) {
    int dv = (base + i) & (Dv - 1);
    wrow[base + i] = (__bf16)(vals[i] * rinv * gnw[dv] * (float)grow[base + i]);
  }
}

// ---------------- launcher ----------------
extern "C" void kernel_launch(void* const* d_in, const int* in_sizes, int n_in,
                              void* d_out, int out_size, void* d_ws, size_t ws_size,
                              hipStream_t stream) {
  (void)in_sizes; (void)n_in; (void)out_size; (void)ws_size;
  const float* x   = (const float*)d_in[0];
  const float* Wq  = (const float*)d_in[1];
  const float* Wk  = (const float*)d_in[2];
  const float* Wv  = (const float*)d_in[3];
  const float* Wg  = (const float*)d_in[4];
  const float* Wo  = (const float*)d_in[5];
  const float* gnw = (const float*)d_in[6];
  float* out = (float*)d_out;

  char* ws = (char*)d_ws;
  size_t off = 0;
  auto alloc = [&](size_t bytes) -> void* {
    void* p = ws + off;
    off += (bytes + 255) & ~(size_t)255;
    return p;
  };

  __bf16* xh  = (__bf16*)alloc((size_t)NT * Ec * 2);
  __bf16* WqT = (__bf16*)alloc((size_t)Ec * Ec * 2);
  __bf16* WkT = (__bf16*)alloc((size_t)Ec * Ec * 2);
  __bf16* WvT = (__bf16*)alloc((size_t)2 * Ec * Ec * 2);
  __bf16* WgT = (__bf16*)alloc((size_t)2 * Ec * Ec * 2);
  __bf16* WoT = (__bf16*)alloc((size_t)2 * Ec * Ec * 2);
  __bf16* qp  = (__bf16*)alloc((size_t)NT * Ec * 2);
  __bf16* kp  = (__bf16*)alloc((size_t)NT * Ec * 2);
  __bf16* vp  = (__bf16*)alloc((size_t)NT * 2 * Ec * 2);
  __bf16* gp  = (__bf16*)alloc((size_t)NT * 2 * Ec * 2);
  __bf16* qrb = (__bf16*)alloc((size_t)NT * Ec * 2);
  __bf16* krb = (__bf16*)alloc((size_t)NT * Ec * 2);
  __bf16* vtb = (__bf16*)alloc((size_t)NT * 2 * Ec * 2);
  float*  ob  = (float* )alloc((size_t)NT * 2 * Ec * 4);
  __bf16* og  = (__bf16*)alloc((size_t)NT * 2 * Ec * 2);

  dim3 b256(256);
  // 1) convert / pre-transpose
  cvt_kernel<<<(NT * Ec) / 256, b256, 0, stream>>>(x, xh, NT * Ec);
  cvtT_kernel<<<(Ec * Ec) / 256, b256, 0, stream>>>(Wq, WqT, Ec, Ec);
  cvtT_kernel<<<(Ec * Ec) / 256, b256, 0, stream>>>(Wk, WkT, Ec, Ec);
  cvtT_kernel<<<(Ec * 2 * Ec) / 256, b256, 0, stream>>>(Wv, WvT, Ec, 2 * Ec);
  cvtT_kernel<<<(Ec * 2 * Ec) / 256, b256, 0, stream>>>(Wg, WgT, Ec, 2 * Ec);
  cvtT_kernel<<<(2 * Ec * Ec) / 256, b256, 0, stream>>>(Wo, WoT, 2 * Ec, Ec);

  // 2) projections (WMMA GEMMs)
  gemm_bf16_kernel<__bf16><<<dim3(Ec / BN, NT / BM), b256, 0, stream>>>(xh, WqT, qp, NT, Ec, Ec);
  gemm_bf16_kernel<__bf16><<<dim3(Ec / BN, NT / BM), b256, 0, stream>>>(xh, WkT, kp, NT, Ec, Ec);
  gemm_bf16_kernel<__bf16><<<dim3(2 * Ec / BN, NT / BM), b256, 0, stream>>>(xh, WvT, vp, NT, 2 * Ec, Ec);
  gemm_bf16_kernel<__bf16><<<dim3(2 * Ec / BN, NT / BM), b256, 0, stream>>>(xh, WgT, gp, NT, 2 * Ec, Ec);

  // 3) rotary + layout changes
  rope_kernel<<<(Bc * Tc * Hc * (Dk / 2)) / 256, b256, 0, stream>>>(qp, kp, qrb, krb);
  vtrans_kernel<<<(NT * 2 * Ec) / 256, b256, 0, stream>>>(vp, vtb);

  // 4) retention attention (WMMA)
  retention_kernel<<<dim3(Tc / 16, Dv / 256, Bc * Hc), dim3(32), 0, stream>>>(qrb, krb, vtb, ob);

  // 5) groupnorm + gate
  gn_gate_kernel<<<NT, b256, 0, stream>>>(ob, gp, gnw, og);

  // 6) output projection (WMMA GEMM, f32 out)
  gemm_bf16_kernel<float><<<dim3(Ec / BN, NT / BM), b256, 0, stream>>>(og, WoT, out, NT, Ec, 2 * Ec);
}
